// NlEncoder_8074538517124
// MI455X (gfx1250) — matrix operand order repaired and checked by reading, not compiled, not run on hardware
//
#include <hip/hip_runtime.h>
#include <hip/hip_bf16.h>

#define BB 8
#define NL 1024
#define DD 128
#define NN 2048
#define NBLK 5
#define NSTEP 3

typedef __bf16 bf16;
typedef __attribute__((ext_vector_type(16))) __bf16 v16bf;
typedef __attribute__((ext_vector_type(8)))  float  v8f;

union FragU { v16bf v; uint4 q[2]; };
union Pack8 { bf16 h[8]; uint4 q; };

static __device__ __forceinline__ v16bf ld_frag(const bf16* lo, const bf16* hi) {
  FragU u;
  u.q[0] = *(const uint4*)lo;
  u.q[1] = *(const uint4*)hi;
  return u.v;
}

static __device__ __forceinline__ v8f wmma_bf16(v16bf a, v16bf b, v8f c) {
  return __builtin_amdgcn_wmma_f32_16x16x32_bf16(false, a, false, b, (short)0, c, false, false);
}

static __device__ __forceinline__ float sigmoidf_(float x) {
  return 1.0f / (1.0f + __expf(-x));
}

// ---------------- prep kernels ----------------

__global__ void k_cvt_adj(const float* __restrict__ a, bf16* __restrict__ o, long n) {
  long i = ((long)blockIdx.x * blockDim.x + threadIdx.x) * 4;
  long stride = (long)gridDim.x * blockDim.x * 4;
  for (long j = i; j < n; j += stride) {
    float4 f = *(const float4*)(a + j);
    Pack8 p;
    p.h[0] = (bf16)f.x; p.h[1] = (bf16)f.y; p.h[2] = (bf16)f.z; p.h[3] = (bf16)f.w;
    *(uint2*)(o + j) = *(uint2*)&p.q;
  }
}

// W: [nmat][K][Ncol] fp32 -> Wt: [nmat][Ncol][K] bf16
__global__ void k_transp_w(const float* __restrict__ W, bf16* __restrict__ Wt,
                           int K, int Ncol, int nmat) {
  int total = nmat * K * Ncol;
  for (int idx = blockIdx.x * blockDim.x + threadIdx.x; idx < total;
       idx += gridDim.x * blockDim.x) {
    int m = idx / (K * Ncol);
    int r = idx % (K * Ncol);
    int k = r / Ncol;
    int n = r % Ncol;
    Wt[((long)m * Ncol + n) * K + k] = (bf16)W[idx];
  }
}

// build x (bf16, row-major) directly into hb
__global__ void k_embed(const int* __restrict__ node, const int* __restrict__ text,
                        const int* __restrict__ lnode,
                        const float* __restrict__ tok, const float* __restrict__ tok1,
                        bf16* __restrict__ xb) {
  int bn = blockIdx.x;
  int b = bn / NN, n = bn % NN, d = threadIdx.x;
  float v;
  if (n < NL) {
    if (d < DD - 1) v = tok[(long)node[b * NL + n] * (DD - 1) + d];
    else            v = (float)text[b * NL + n];
  } else {
    v = tok1[(long)lnode[b * NL + (n - NL)] * DD + d];
  }
  xb[((long)b * NN + n) * DD + d] = (bf16)v;
}

// ---------------- h = x @ in_W + b ----------------
// block: 16 rows x 128 cols, 8 waves (one 16x16 tile each). In-place safe.
__global__ void k_inproj(const bf16* __restrict__ xb, const bf16* __restrict__ Wt,
                         const float* __restrict__ bias,
                         float* __restrict__ h32, bf16* __restrict__ hb,
                         bf16* __restrict__ hT) {
  int blk = blockIdx.x;
  int b = blk >> 7, rt = blk & 127;
  int row0 = rt * 16;
  int wave = threadIdx.x >> 5, lane = threadIdx.x & 31;
  int half = lane >> 4, nl = lane & 15;
  int col = wave * 16 + nl;
  const bf16* xrow = xb + ((long)b * NN + row0 + nl) * DD;
  const bf16* wtc  = Wt + (long)col * DD;
  v8f acc = {};
#pragma unroll
  for (int k0 = 0; k0 < DD; k0 += 32) {
    v16bf a = ld_frag(xrow + k0 + half * 8, xrow + k0 + 16 + half * 8);
    v16bf w = ld_frag(wtc + k0 + half * 8, wtc + k0 + 16 + half * 8);
    acc = wmma_bf16(a, w, acc);
  }
  float bv = bias[col];
  __syncthreads();           // all reads of xb done before in-place overwrite
  Pack8 pk;
#pragma unroll
  for (int r = 0; r < 8; ++r) {
    float v = acc[r] + bv;
    long idx = ((long)b * NN + row0 + 8 * half + r) * DD + col;
    h32[idx] = v;
    hb[idx]  = (bf16)v;
    pk.h[r]  = (bf16)v;
  }
  *(uint4*)(hT + ((long)b * DD + col) * NN + row0 + 8 * half) = pk.q;
}

// ---------------- msgs = adj @ h ----------------
// 2x2 register tiling: each wave computes a 32x32 output patch (4 WMMA tiles),
// reusing each A/B fragment twice -> 16 FLOP/byte from L2.
// Block = 8 waves = 4 row-positions x 2 col-positions = 128 rows x 64 cols.
// Grid = BB * 16 row-supergroups * 2 col-groups = 256 blocks.
__global__ void k_msgs(const bf16* __restrict__ adj, const bf16* __restrict__ hT,
                       bf16* __restrict__ msgs) {
  int blk = blockIdx.x;
  int b = blk >> 5;
  int rem = blk & 31;
  int cg = rem >> 4, rg = rem & 15;
  int wave = threadIdx.x >> 5, lane = threadIdx.x & 31;
  int half = lane >> 4, nl = lane & 15;
  int wr = wave & 3, wc = wave >> 2;
  int rbase = rg * 128 + wr * 32;       // 32 rows for this wave
  int cbase = cg * 64 + wc * 32;        // 32 cols for this wave

  const bf16* arow0 = adj + ((long)b * NN + rbase + nl) * NN;
  const bf16* arow1 = adj + ((long)b * NN + rbase + 16 + nl) * NN;
  const bf16* bcol0 = hT + ((long)b * DD + cbase + nl) * NN;
  const bf16* bcol1 = hT + ((long)b * DD + cbase + 16 + nl) * NN;

  v8f acc00 = {}, acc01 = {}, acc10 = {}, acc11 = {};
#pragma unroll 2
  for (int k0 = 0; k0 < NN; k0 += 32) {
    v16bf a0 = ld_frag(arow0 + k0 + half * 8, arow0 + k0 + 16 + half * 8);
    v16bf a1 = ld_frag(arow1 + k0 + half * 8, arow1 + k0 + 16 + half * 8);
    v16bf b0 = ld_frag(bcol0 + k0 + half * 8, bcol0 + k0 + 16 + half * 8);
    v16bf b1 = ld_frag(bcol1 + k0 + half * 8, bcol1 + k0 + 16 + half * 8);
    acc00 = wmma_bf16(a0, b0, acc00);
    acc01 = wmma_bf16(a0, b1, acc01);
    acc10 = wmma_bf16(a1, b0, acc10);
    acc11 = wmma_bf16(a1, b1, acc11);
  }
#pragma unroll
  for (int r = 0; r < 8; ++r) {
    long row_a = (long)b * NN + rbase + 8 * half + r;
    long row_b = row_a + 16;
    msgs[row_a * DD + cbase + nl]      = (bf16)acc00[r];
    msgs[row_a * DD + cbase + 16 + nl] = (bf16)acc01[r];
    msgs[row_b * DD + cbase + nl]      = (bf16)acc10[r];
    msgs[row_b * DD + cbase + 16 + nl] = (bf16)acc11[r];
  }
}

// ---------------- fused GRU step ----------------
__global__ void k_gate(bf16* __restrict__ hb, float* __restrict__ h32,
                       const bf16* __restrict__ msgs,
                       const bf16* __restrict__ ugWt, const float* __restrict__ ugb,
                       const bf16* __restrict__ rgWt, const float* __restrict__ rgb,
                       const bf16* __restrict__ htWt, const float* __restrict__ htb,
                       bf16* __restrict__ hT) {
  __shared__ __align__(16) bf16 rhs[16 * DD];
  int blk = blockIdx.x;
  int b = blk >> 7, rt = blk & 127;
  int row0 = rt * 16;
  int wave = threadIdx.x >> 5, lane = threadIdx.x & 31;
  int half = lane >> 4, nl = lane & 15;
  int col = wave * 16 + nl;
  const bf16* hrow = hb + ((long)b * NN + row0 + nl) * DD;
  const bf16* mrow = msgs + ((long)b * NN + row0 + nl) * DD;
  const bf16* ugc = ugWt + (long)col * 256;
  const bf16* rgc = rgWt + (long)col * 256;
  const bf16* htc = htWt + (long)col * 256;

  v8f az = {}, ar = {};
#pragma unroll
  for (int k0 = 0; k0 < DD; k0 += 32) {  // comb[:,:128] = h
    v16bf a = ld_frag(hrow + k0 + half * 8, hrow + k0 + 16 + half * 8);
    v16bf wz = ld_frag(ugc + k0 + half * 8, ugc + k0 + 16 + half * 8);
    v16bf wr = ld_frag(rgc + k0 + half * 8, rgc + k0 + 16 + half * 8);
    az = wmma_bf16(a, wz, az);
    ar = wmma_bf16(a, wr, ar);
  }
#pragma unroll
  for (int k0 = 0; k0 < DD; k0 += 32) {  // comb[:,128:] = msgs
    v16bf a = ld_frag(mrow + k0 + half * 8, mrow + k0 + 16 + half * 8);
    v16bf wz = ld_frag(ugc + 128 + k0 + half * 8, ugc + 128 + k0 + 16 + half * 8);
    v16bf wr = ld_frag(rgc + 128 + k0 + half * 8, rgc + 128 + k0 + 16 + half * 8);
    az = wmma_bf16(a, wz, az);
    ar = wmma_bf16(a, wr, ar);
  }
  float zb = ugb[col], rb = rgb[col], cb = htb[col];
  float zv[8], hold[8];
#pragma unroll
  for (int r = 0; r < 8; ++r) {
    long idx = ((long)b * NN + row0 + 8 * half + r) * DD + col;
    hold[r] = h32[idx];
    zv[r] = sigmoidf_(az[r] + zb);
    float rv = sigmoidf_(ar[r] + rb);
    rhs[(8 * half + r) * DD + col] = (bf16)(rv * hold[r]);
  }
  __syncthreads();  // rh tile ready; all hb/h32 reads complete

  v8f ah = {};
  const bf16* rhrow = rhs + nl * DD;
#pragma unroll
  for (int k0 = 0; k0 < DD; k0 += 32) {  // comb2[:,:128] = r*h (from LDS)
    v16bf a = ld_frag(rhrow + k0 + half * 8, rhrow + k0 + 16 + half * 8);
    v16bf wh = ld_frag(htc + k0 + half * 8, htc + k0 + 16 + half * 8);
    ah = wmma_bf16(a, wh, ah);
  }
#pragma unroll
  for (int k0 = 0; k0 < DD; k0 += 32) {  // comb2[:,128:] = msgs
    v16bf a = ld_frag(mrow + k0 + half * 8, mrow + k0 + 16 + half * 8);
    v16bf wh = ld_frag(htc + 128 + k0 + half * 8, htc + 128 + k0 + 16 + half * 8);
    ah = wmma_bf16(a, wh, ah);
  }
  Pack8 pk;
#pragma unroll
  for (int r = 0; r < 8; ++r) {
    float cand = tanhf(ah[r] + cb);
    float v = zv[r] * hold[r] + (1.0f - zv[r]) * cand;
    long idx = ((long)b * NN + row0 + 8 * half + r) * DD + col;
    h32[idx] = v;
    hb[idx]  = (bf16)v;
    pk.h[r]  = (bf16)v;
  }
  *(uint4*)(hT + ((long)b * DD + col) * NN + row0 + 8 * half) = pk.q;
}

// ---------------- head: logits -> softmax -> loss ----------------
__global__ void k_head(const float* __restrict__ h32, const int* __restrict__ node,
                       const float* __restrict__ res, const float* __restrict__ w2,
                       const float* __restrict__ b2, float* __restrict__ out) {
  __shared__ float red[NL];
  int b = blockIdx.x, n = threadIdx.x;
  const float* hr = h32 + ((long)b * NN + n) * DD;
  float dot = b2[0];
#pragma unroll 8
  for (int d = 0; d < DD; ++d) dot += hr[d] * w2[d];
  bool mask = (node[b * NL + n] == 2);
  float logit = mask ? dot : -1.0e9f;
  red[n] = logit; __syncthreads();
  for (int s = NL / 2; s > 0; s >>= 1) {
    if (n < s) red[n] = fmaxf(red[n], red[n + s]);
    __syncthreads();
  }
  float mx = red[0]; __syncthreads();
  float e = __expf(logit - mx);
  red[n] = e; __syncthreads();
  for (int s = NL / 2; s > 0; s >>= 1) {
    if (n < s) red[n] += red[n + s];
    __syncthreads();
  }
  float sum = red[0]; __syncthreads();
  float sm = e / sum;
  out[8 + b * NL + n] = sm;
  float c = fminf(fmaxf(sm, 1e-10f), 1.0f);
  red[n] = -__logf(c) * res[b * NL + n]; __syncthreads();
  for (int s = NL / 2; s > 0; s >>= 1) {
    if (n < s) red[n] += red[n + s];
    __syncthreads();
  }
  if (n == 0) out[b] = red[0];
}

__global__ void k_copyx(const float* __restrict__ h32, float* __restrict__ out) {
  long i = (long)blockIdx.x * blockDim.x + threadIdx.x;
  if (i < (long)BB * NL * DD) {
    long b = i / ((long)NL * DD);
    long r = i % ((long)NL * DD);
    out[8 + BB * NL + i] = h32[b * (long)NN * DD + r];
  }
}

// ---------------- host launcher ----------------
extern "C" void kernel_launch(void* const* d_in, const int* in_sizes, int n_in,
                              void* d_out, int out_size, void* d_ws, size_t ws_size,
                              hipStream_t stream) {
  (void)in_sizes; (void)n_in; (void)out_size; (void)ws_size;
  const int*   input_node = (const int*)d_in[0];
  const float* inputad    = (const float*)d_in[2];
  const float* res        = (const float*)d_in[3];
  const int*   inputtext  = (const int*)d_in[4];
  const int*   linenode   = (const int*)d_in[5];
  const float* tok_emb    = (const float*)d_in[8];
  const float* tok_emb1   = (const float*)d_in[9];
  const float* in_W  = (const float*)d_in[10];
  const float* in_b  = (const float*)d_in[11];
  const float* ug_W  = (const float*)d_in[12];
  const float* ug_b  = (const float*)d_in[13];
  const float* rg_W  = (const float*)d_in[14];
  const float* rg_b  = (const float*)d_in[15];
  const float* ht_W  = (const float*)d_in[16];
  const float* ht_b  = (const float*)d_in[17];
  const float* res2_W = (const float*)d_in[18];
  const float* res2_b = (const float*)d_in[19];

  size_t off = 0;
  auto carve = [&](size_t bytes) -> void* {
    void* p = (char*)d_ws + off;
    off += (bytes + 255) & ~(size_t)255;
    return p;
  };
  bf16* adjb  = (bf16*)carve((size_t)BB * NN * NN * 2);   // 67 MB (L2-resident)
  float* h32  = (float*)carve((size_t)BB * NN * DD * 4);
  bf16* hb    = (bf16*)carve((size_t)BB * NN * DD * 2);
  bf16* hT    = (bf16*)carve((size_t)BB * DD * NN * 2);
  bf16* msgsb = (bf16*)carve((size_t)BB * NN * DD * 2);
  bf16* inWt  = (bf16*)carve((size_t)NBLK * DD * DD * 2);
  bf16* ugWt  = (bf16*)carve((size_t)NBLK * DD * 256 * 2);
  bf16* rgWt  = (bf16*)carve((size_t)NBLK * DD * 256 * 2);
  bf16* htWt  = (bf16*)carve((size_t)NBLK * DD * 256 * 2);

  // prep
  k_cvt_adj<<<8192, 256, 0, stream>>>(inputad, adjb, (long)BB * NN * NN);
  k_transp_w<<<320, 256, 0, stream>>>(in_W, inWt, DD, DD, NBLK);
  k_transp_w<<<640, 256, 0, stream>>>(ug_W, ugWt, 256, DD, NBLK);
  k_transp_w<<<640, 256, 0, stream>>>(rg_W, rgWt, 256, DD, NBLK);
  k_transp_w<<<640, 256, 0, stream>>>(ht_W, htWt, 256, DD, NBLK);
  k_embed<<<BB * NN, DD, 0, stream>>>(input_node, inputtext, linenode,
                                      tok_emb, tok_emb1, hb);

  for (int i = 0; i < NBLK; ++i) {
    k_inproj<<<BB * 128, 256, 0, stream>>>(hb, inWt + (size_t)i * DD * DD,
                                           in_b + i * DD, h32, hb, hT);
    for (int s = 0; s < NSTEP; ++s) {
      k_msgs<<<BB * 32, 256, 0, stream>>>(adjb, hT, msgsb);
      k_gate<<<BB * 128, 256, 0, stream>>>(hb, h32, msgsb,
          ugWt + (size_t)i * DD * 256, ug_b + i * DD,
          rgWt + (size_t)i * DD * 256, rg_b + i * DD,
          htWt + (size_t)i * DD * 256, ht_b + i * DD,
          hT);
    }
  }

  k_head<<<BB, NL, 0, stream>>>(h32, input_node, res, res2_W, res2_b, (float*)d_out);
  k_copyx<<<4096, 256, 0, stream>>>(h32, (float*)d_out);
}
